// DeltaBlock_75265006895518
// MI455X (gfx1250) — compile-verified, matrix operand order checked by reading, not run on hardware
//
#include <hip/hip_runtime.h>
#include <hip/hip_bf16.h>
#include <math.h>

// Problem constants (match reference)
constexpr int Bc = 4, Tc = 4096, Dc = 1024, Hc = 16, HDc = 64, Cc = 64, Nc = 64;

typedef __attribute__((ext_vector_type(16))) __bf16 bf16x16;
typedef __attribute__((ext_vector_type(8)))  float  floatx8;
typedef unsigned int u32x4 __attribute__((ext_vector_type(4)));
typedef int          i32x4 __attribute__((ext_vector_type(4)));
typedef int          i32x8 __attribute__((ext_vector_type(8)));

// ---------------------------------------------------------------------------
// WMMA: D(16x16,f32) = A(16x32,bf16) * B(32x16,bf16) + C
// ---------------------------------------------------------------------------
__device__ inline floatx8 wmma_bf16(bf16x16 a, bf16x16 b, floatx8 c) {
  return __builtin_amdgcn_wmma_f32_16x16x32_bf16(false, a, false, b,
                                                 (short)0, c, false, false);
}

// Fragment-major ("A-order"/"B-order") indexing. A 64x64 tile is stored as
// 8 fragments x 32 lanes x 16 bf16, so a fragment load is one 32B LDS read.
// A-order (operand A, row-major MxK; also serves B^T): frag=(m>>4)*2+(k>>5)
__device__ inline int aord64(int m, int k) {
  return (((((m >> 4) << 1) + (k >> 5)) << 5) + (m & 15) + (((k >> 3) & 1) << 4)) * 16 +
         ((k & 7) | ((k & 16) >> 1));
}
// B-order (operand B, row-major KxN): frag=(k>>5)*4+(n>>4)
__device__ inline int bord64(int k, int n) {
  return (((((k >> 5) << 2) + (n >> 4)) << 5) + (n & 15) + (((k >> 3) & 1) << 4)) * 16 +
         ((k & 7) | ((k & 16) >> 1));
}
// One fragment = contiguous 32B per lane -> ds_load_b128 x2
__device__ inline bf16x16 frag_ld(const __bf16* t, int fragIdx, int lane) {
  return *(const bf16x16*)(t + (((fragIdx << 5) + lane) << 4));
}
// C/D map (32-bit 16x16): lane = (M>=8?16:0)+N, vgpr e -> M = m0+e(+8 hi lanes)
__device__ inline floatx8 load_c_f(const float* t, int ld, int m0, int n0, int lane) {
  int n  = n0 + (lane & 15);
  int mb = m0 + ((lane >> 4) << 3);
  floatx8 c;
#pragma unroll
  for (int e = 0; e < 8; ++e) c[e] = t[(mb + e) * ld + n];
  return c;
}
__device__ inline void store_d_f(float* t, int ld, int m0, int n0, int lane, floatx8 d) {
  int n  = n0 + (lane & 15);
  int mb = m0 + ((lane >> 4) << 3);
#pragma unroll
  for (int e = 0; e < 8; ++e) t[(mb + e) * ld + n] = d[e];
}

// ---------------------------------------------------------------------------
// Kernel 1/6: big GEMM  O = A(MxK) @ W(KxN) [+ R]; bf16 WMMA, frag-major LDS,
// K-step 64: 4 WMMAs per wave per barrier pair.
// ---------------------------------------------------------------------------
__global__ __launch_bounds__(256) void gemm_bf16_kernel(
    const float* __restrict__ A, const float* __restrict__ W,
    const float* __restrict__ R, float* __restrict__ O,
    int M, int N, int K) {
  __shared__ __attribute__((aligned(32))) __bf16 As[64 * 64];  // A-order
  __shared__ __attribute__((aligned(32))) __bf16 Ws[64 * 64];  // B-order
  int tid = threadIdx.x, lane = tid & 31, wv = tid >> 5;
  int bm = blockIdx.y * 64, bn = blockIdx.x * 64;
  int mi4 = wv >> 1;              // A row-block (0..3)
  int nb0 = (wv & 1) * 2;         // first of two B n-blocks
  floatx8 acc0 = {}, acc1 = {};
  for (int k0 = 0; k0 < K; k0 += 64) {
#pragma unroll
    for (int l = 0; l < 4; ++l) {           // A tile: 64x64 f32 as float4 loads
      int e4 = tid + l * 256;
      int r = e4 >> 4, c4 = (e4 & 15) * 4;
      float4 v = *(const float4*)&A[(size_t)(bm + r) * K + k0 + c4];
      As[aord64(r, c4 + 0)] = (__bf16)v.x;
      As[aord64(r, c4 + 1)] = (__bf16)v.y;
      As[aord64(r, c4 + 2)] = (__bf16)v.z;
      As[aord64(r, c4 + 3)] = (__bf16)v.w;
    }
#pragma unroll
    for (int l = 0; l < 4; ++l) {           // W tile: 64x64, 4-row column segs
      int nn = tid & 63;
      int r4 = ((tid >> 6) + l * 4) * 4;
      float a0 = W[(size_t)(k0 + r4 + 0) * N + bn + nn];
      float a1 = W[(size_t)(k0 + r4 + 1) * N + bn + nn];
      float a2 = W[(size_t)(k0 + r4 + 2) * N + bn + nn];
      float a3 = W[(size_t)(k0 + r4 + 3) * N + bn + nn];
      Ws[bord64(r4 + 0, nn)] = (__bf16)a0;
      Ws[bord64(r4 + 1, nn)] = (__bf16)a1;
      Ws[bord64(r4 + 2, nn)] = (__bf16)a2;
      Ws[bord64(r4 + 3, nn)] = (__bf16)a3;
    }
    __syncthreads();
    if (k0 + 64 < K) {  // global_prefetch_b8 for next tiles
      __builtin_prefetch(&A[(size_t)(bm + (tid >> 2)) * K + k0 + 64], 0, 3);
      __builtin_prefetch(&W[(size_t)(k0 + 64 + (tid >> 3)) * N + bn], 0, 3);
    }
    bf16x16 a0  = frag_ld(As, mi4 * 2 + 0, lane);
    bf16x16 a1  = frag_ld(As, mi4 * 2 + 1, lane);
    bf16x16 b00 = frag_ld(Ws, 0 + nb0 + 0, lane);
    bf16x16 b01 = frag_ld(Ws, 4 + nb0 + 0, lane);
    bf16x16 b10 = frag_ld(Ws, 0 + nb0 + 1, lane);
    bf16x16 b11 = frag_ld(Ws, 4 + nb0 + 1, lane);
    acc0 = wmma_bf16(a0, b00, acc0);
    acc0 = wmma_bf16(a1, b01, acc0);
    acc1 = wmma_bf16(a0, b10, acc1);
    acc1 = wmma_bf16(a1, b11, acc1);
    __syncthreads();
  }
  int mi = mi4 * 16, ni0 = nb0 * 16;
#pragma unroll
  for (int j = 0; j < 2; ++j) {
    floatx8 acc = j ? acc1 : acc0;
    int n  = bn + ni0 + j * 16 + (lane & 15);
    int mb = bm + mi + ((lane >> 4) << 3);
#pragma unroll
    for (int e = 0; e < 8; ++e) {
      float v = acc[e];
      if (R) v += R[(size_t)(mb + e) * N + n];
      O[(size_t)(mb + e) * N + n] = v;
    }
  }
}

// ---------------------------------------------------------------------------
// Kernel 2/6: skinny projections gate/beta/decay — VALU dots.
// ---------------------------------------------------------------------------
__global__ __launch_bounds__(256) void smallproj_kernel(
    const float* __restrict__ x, const float* __restrict__ wg,
    const float* __restrict__ wb, const float* __restrict__ wa,
    const float* __restrict__ dtb, const float* __restrict__ Alog,
    float* __restrict__ gate, float* __restrict__ beta, float* __restrict__ decay) {
  size_t idx = (size_t)blockIdx.x * 256 + threadIdx.x;
  if (idx >= (size_t)Bc * Tc * Hc) return;
  size_t bt = idx / Hc;
  int h = (int)(idx % Hc);
  const float* xr = x + bt * Dc;
  float sg = 0.f, sb = 0.f, sa = 0.f;
  for (int k = 0; k < Dc; ++k) {
    float v = xr[k];
    sg += v * wg[(size_t)k * Hc + h];
    sb += v * wb[(size_t)k * Hc + h];
    sa += v * wa[(size_t)k * Hc + h];
  }
  gate[idx] = 1.f / (1.f + expf(-sg));
  beta[idx] = 1.f / (1.f + expf(-sb));
  float z = sa + dtb[h];
  float sp = (z > 20.f) ? z : log1pf(expf(z));
  decay[idx] = -expf(Alog[h]) * sp;
}

// ---------------------------------------------------------------------------
// Kernel 3/6: prep — normalize rk, shifted wk, v*beta, wk*beta, chunk cumsum.
// ---------------------------------------------------------------------------
__global__ __launch_bounds__(64) void prep_kernel(
    const float* __restrict__ x, const float* __restrict__ xv,
    const float* __restrict__ beta, const float* __restrict__ decay,
    float* __restrict__ rk, float* __restrict__ wk,
    float* __restrict__ vb, float* __restrict__ wkb, float* __restrict__ gcum) {
  int blk = blockIdx.x;
  int b = blk / (Hc * Nc), rem = blk % (Hc * Nc), h = rem / Nc, n = rem % Nc;
  int c = threadIdx.x, t = n * Cc + c;
  const float* xr = x + ((size_t)(b * Tc + t)) * Dc + h * HDc;
  float s = 0.f;
  for (int i = 0; i < HDc; ++i) { float v = xr[i]; s += v * v; }
  float inv = 1.f / fmaxf(sqrtf(s), 1e-12f);
  float bta = beta[((size_t)(b * Tc + t)) * Hc + h];
  size_t ob = (((size_t)(b * Hc + h) * Tc) + t) * HDc;
  for (int i = 0; i < HDc; ++i) rk[ob + i] = xr[i] * inv;
  if (t == 0) {
    for (int i = 0; i < HDc; ++i) { wk[ob + i] = 0.f; wkb[ob + i] = 0.f; }
  } else {
    const float* xp = x + ((size_t)(b * Tc + t - 1)) * Dc + h * HDc;
    float s2 = 0.f;
    for (int i = 0; i < HDc; ++i) { float v = xp[i]; s2 += v * v; }
    float inv2 = 1.f / fmaxf(sqrtf(s2), 1e-12f);
    for (int i = 0; i < HDc; ++i) {
      float w = xp[i] * inv2;
      wk[ob + i] = w;
      wkb[ob + i] = w * bta;
    }
  }
  const float* vr = xv + ((size_t)(b * Tc + t)) * Dc + h * HDc;
  for (int i = 0; i < HDc; ++i) vb[ob + i] = vr[i] * bta;
  __shared__ float g[Cc];
  g[c] = decay[((size_t)(b * Tc + t)) * Hc + h];
  __syncthreads();
  float cs = 0.f;
  for (int j = 0; j <= c; ++j) cs += g[j];
  gcum[((size_t)(b * Hc + h) * Tc) + t] = cs;
}

// ---------------------------------------------------------------------------
// Kernel 4/6: chunk-local WY transform. Emits, per (b,h,n):
//   vcorr (f32), and fragment-major bf16 tiles for the scan:
//   t0g=-wk_cd (A-ord), t1g=rk*dexp (A-ord), t2g=intra*L (A-ord),
//   t3g=(wk*dw)^T (A-ord of the transposed d x C matrix).
// ---------------------------------------------------------------------------
__global__ __launch_bounds__(256) void chunk_kernel(
    const float* __restrict__ rk, const float* __restrict__ wk,
    const float* __restrict__ vb, const float* __restrict__ wkb,
    const float* __restrict__ gcum, float* __restrict__ vcorr,
    __bf16* __restrict__ t0g, __bf16* __restrict__ t1g,
    __bf16* __restrict__ t2g, __bf16* __restrict__ t3g) {
  __shared__ __attribute__((aligned(32))) char smemC[6 * 8192 + 512];
  __bf16* wkbT = (__bf16*)(smemC);            // A-order wkb, later B-order wkb*dexp
  __bf16* wkA  = (__bf16*)(smemC + 8192);     // A-order wk (serves B^T operand)
  __bf16* rkA  = (__bf16*)(smemC + 16384);    // A-order rk
  __bf16* vbB  = (__bf16*)(smemC + 24576);    // B-order v*beta
  __bf16* Bm   = (__bf16*)(smemC + 32768);    // row-major strict-lower matrix
  __bf16* Abf  = (__bf16*)(smemC + 40960);    // A-order WY matrix A
  float* cum_s  = (float*)(smemC + 49152);
  float* dexp_s = cum_s + 64;

  int blk = blockIdx.x;
  int b = blk / (Hc * Nc), rem = blk % (Hc * Nc), h = rem / Nc, n = rem % Nc;
  int tid = threadIdx.x, lane = tid & 31, wv = tid >> 5;
  size_t tix  = (size_t)(b * Hc + h) * Nc + n;
  size_t base = tix * 4096;
  size_t toff = tix * 4096;

  // stage tiles into fragment-major LDS (global loads coalesced)
#pragma unroll
  for (int i = 0; i < 16; ++i) {
    int lin = tid + i * 256, m = lin >> 6, k = lin & 63;
    wkbT[aord64(m, k)] = (__bf16)wkb[base + lin];
    wkA [aord64(m, k)] = (__bf16)wk [base + lin];
    rkA [aord64(m, k)] = (__bf16)rk [base + lin];
    vbB [bord64(m, k)] = (__bf16)vb [base + lin];
  }
  if (tid < Cc) {
    float cu = gcum[(size_t)(b * Hc + h) * Tc + n * Cc + tid];
    cum_s[tid]  = cu;
    dexp_s[tid] = expf(cu);
  }
  __syncthreads();

  int mi = (wv >> 1) * 16;
  int nh = (wv & 1);
  float lastc = cum_s[Cc - 1];

  // Phase 1: kk = wkb @ wk^T (-> Bm), intra = (rk @ wk^T)*L (-> t2g)
  {
    bf16x16 awb0 = frag_ld(wkbT, (mi >> 4) * 2 + 0, lane);
    bf16x16 awb1 = frag_ld(wkbT, (mi >> 4) * 2 + 1, lane);
    bf16x16 ark0 = frag_ld(rkA,  (mi >> 4) * 2 + 0, lane);
    bf16x16 ark1 = frag_ld(rkA,  (mi >> 4) * 2 + 1, lane);
#pragma unroll
    for (int jj = 0; jj < 2; ++jj) {
      int ni = (nh * 2 + jj) * 16;
      bf16x16 bw0 = frag_ld(wkA, (ni >> 4) * 2 + 0, lane);  // wk^T
      bf16x16 bw1 = frag_ld(wkA, (ni >> 4) * 2 + 1, lane);
      floatx8 kkacc = {}, iacc = {};
      kkacc = wmma_bf16(awb0, bw0, kkacc);
      kkacc = wmma_bf16(awb1, bw1, kkacc);
      iacc  = wmma_bf16(ark0, bw0, iacc);
      iacc  = wmma_bf16(ark1, bw1, iacc);
      int ncol = ni + (lane & 15), mb = mi + ((lane >> 4) << 3);
#pragma unroll
      for (int e = 0; e < 8; ++e) {
        int m = mb + e;
        float bmv = (m > ncol) ? -kkacc[e] * expf(cum_s[m] - cum_s[ncol]) : 0.f;
        Bm[m * 64 + ncol] = (__bf16)bmv;
        float L = (m >= ncol) ? expf(cum_s[m] - cum_s[ncol]) : 0.f;
        t2g[toff + aord64(m, ncol)] = (__bf16)(iacc[e] * L);
      }
    }
  }
  __syncthreads();

  // Phase 2a: rebuild wkbT region as B-order wkb*dexp (re-read global)
#pragma unroll
  for (int i = 0; i < 16; ++i) {
    int lin = tid + i * 256, m = lin >> 6, k = lin & 63;
    wkbT[bord64(m, k)] = (__bf16)(wkb[base + lin] * dexp_s[m]);
  }
  // Phase 2b: emit t1g = rk*dexp, t3g = (wk*dw)^T (from staged LDS)
#pragma unroll
  for (int i = 0; i < 16; ++i) {
    int lin = tid + i * 256, m = lin >> 6, k = lin & 63;
    t1g[toff + aord64(m, k)] = (__bf16)((float)rkA[aord64(m, k)] * dexp_s[m]);
    float dwk = expf(lastc - cum_s[k]);
    t3g[toff + aord64(m, k)] = (__bf16)((float)wkA[aord64(k, m)] * dwk);
  }
  // Phase 2c: forward substitution A = (I - Bm)^-1 directly into A-order bf16
  if (tid < Cc) {
    int k = tid;
    for (int i = 0; i < Cc; ++i) {
      float s = (i == k) ? 1.f : 0.f;
      for (int j = 0; j < i; ++j)
        s += (float)Bm[i * 64 + j] * (float)Abf[aord64(j, k)];
      Abf[aord64(i, k)] = (__bf16)s;
    }
  }
  __syncthreads();

  // Phase 3: v_corr = A @ v (f32 out), wk_cd = A @ (wkb*dexp) (-> -t0g)
  {
    bf16x16 aA0 = frag_ld(Abf, (mi >> 4) * 2 + 0, lane);
    bf16x16 aA1 = frag_ld(Abf, (mi >> 4) * 2 + 1, lane);
#pragma unroll
    for (int jj = 0; jj < 2; ++jj) {
      int ni = (nh * 2 + jj) * 16;
      floatx8 a1 = {}, a2 = {};
      a1 = wmma_bf16(aA0, frag_ld(vbB,  0 + (ni >> 4), lane), a1);
      a1 = wmma_bf16(aA1, frag_ld(vbB,  4 + (ni >> 4), lane), a1);
      a2 = wmma_bf16(aA0, frag_ld(wkbT, 0 + (ni >> 4), lane), a2);
      a2 = wmma_bf16(aA1, frag_ld(wkbT, 4 + (ni >> 4), lane), a2);
      int ncol = ni + (lane & 15), mb = mi + ((lane >> 4) << 3);
#pragma unroll
      for (int e = 0; e < 8; ++e) {
        int m = mb + e;
        vcorr[base + (size_t)m * 64 + ncol] = a1[e];
        t0g[toff + aord64(m, ncol)] = (__bf16)(-a2[e]);
      }
    }
  }
}

// ---------------------------------------------------------------------------
// Kernel 5/6: sequential chunk scan; S in LDS; per-chunk operand tiles are
// verbatim 8KB fragment-major copies -> TENSOR_LOAD_TO_LDS (TDM), one wave
// per tile, s_wait_tensorcnt, then workgroup barrier.
// ---------------------------------------------------------------------------
__global__ __launch_bounds__(256) void scan_kernel(
    const float* __restrict__ vcorr,
    const __bf16* __restrict__ t0g, const __bf16* __restrict__ t1g,
    const __bf16* __restrict__ t2g, const __bf16* __restrict__ t3g,
    const float* __restrict__ gcum, const float* __restrict__ gate,
    float* __restrict__ og) {
  __shared__ __attribute__((aligned(32))) char smemS[16384 + 8192 + 4 * 8192];
  float*  S    = (float*)(smemS);             // 64x64 f32 state
  __bf16* Sb   = (__bf16*)(smemS + 16384);    // B-order bf16 copy of S
  __bf16* tl0  = (__bf16*)(smemS + 24576);    // -wk_cd   (A-order)
  __bf16* tl1  = (__bf16*)(smemS + 32768);    // rk*dexp  (A-order)
  __bf16* tl2  = (__bf16*)(smemS + 40960);    // intra*L  (A-order)
  __bf16* tl3  = (__bf16*)(smemS + 49152);    // (wk*dw)^T(A-order)
  __bf16* vnb  = tl0;                         // alias: v_new (B-order) reuses tl0
  float*  part = (float*)(smemS + 32768);     // alias: reduction reuses tl1

  int b = blockIdx.x / Hc, h = blockIdx.x % Hc;
  int tid = threadIdx.x, lane = tid & 31, wv = tid >> 5;
  int mi = (wv >> 1) * 16;
  int nh = (wv & 1);
#pragma unroll
  for (int i = 0; i < 16; ++i) S[tid + i * 256] = 0.f;
#pragma unroll
  for (int i = 0; i < 16; ++i) Sb[tid + i * 256] = (__bf16)0.f;
  __syncthreads();

  for (int n = 0; n < Nc; ++n) {
    size_t tix  = (size_t)(b * Hc + h) * Nc + n;
    size_t base = tix * 4096;
    // ---- stage 4 operand tiles: TDM async tensor load, one wave per tile
    if (wv < 4) {
      const __bf16* gsrc = (wv == 0) ? (t0g + base)
                         : (wv == 1) ? (t1g + base)
                         : (wv == 2) ? (t2g + base)
                                     : (t3g + base);
      unsigned int ldsoff = (unsigned int)(size_t)(smemS + 24576 + wv * 8192);
      unsigned long long ga = (unsigned long long)(size_t)gsrc;
      u32x4 g0;
      g0[0] = 1u;                                   // count=1
      g0[1] = ldsoff;                               // lds_addr
      g0[2] = (unsigned int)ga;                     // global_addr lo
      g0[3] = (unsigned int)((ga >> 32) & 0x1FFFFFFu) | (2u << 30);  // hi | type=2
      i32x8 g1;
      g1[0] = 0x00030000;            // data_size=3 (8B units)
      g1[1] = (int)(1024u << 16);    // tensor_dim0 = 1024 (low16)
      g1[2] = 0x00010000;            // tensor_dim1 = 1
      g1[3] = (int)(1024u << 16);    // tile_dim0 = 1024
      g1[4] = 0;                     // tile_dim1/2 unused
      g1[5] = 1024;                  // tensor_dim0_stride lo
      g1[6] = 0; g1[7] = 0;
      i32x4 z4 = {0, 0, 0, 0};
#if __clang_major__ >= 23
      i32x8 z8 = {0, 0, 0, 0, 0, 0, 0, 0};
      __builtin_amdgcn_tensor_load_to_lds(g0, g1, z4, z4, z8, 0);
#else
      __builtin_amdgcn_tensor_load_to_lds(g0, g1, z4, z4, 0);
#endif
      __builtin_amdgcn_s_wait_tensorcnt(0);
    }
    __syncthreads();

    float last  = gcum[(size_t)(b * Hc + h) * Tc + n * Cc + 63];
    float elast = expf(last);

    // ---- v_new = vcorr + (-wk_cd) @ S  (frags kept in regs across barrier)
    floatx8 vn[2];
#pragma unroll
    for (int jj = 0; jj < 2; ++jj) {
      int ni = (nh * 2 + jj) * 16;
      floatx8 acc = load_c_f(vcorr + base, 64, mi, ni, lane);
      acc = wmma_bf16(frag_ld(tl0, (mi >> 4) * 2 + 0, lane),
                      frag_ld(Sb, 0 + (ni >> 4), lane), acc);
      acc = wmma_bf16(frag_ld(tl0, (mi >> 4) * 2 + 1, lane),
                      frag_ld(Sb, 4 + (ni >> 4), lane), acc);
      vn[jj] = acc;
    }
    __syncthreads();   // all tl0 reads done; reuse region as v_new (B-order)
#pragma unroll
    for (int jj = 0; jj < 2; ++jj) {
      int ni = (nh * 2 + jj) * 16;
      int ncol = ni + (lane & 15), mb = mi + ((lane >> 4) << 3);
#pragma unroll
      for (int e = 0; e < 8; ++e) vnb[bord64(mb + e, ncol)] = (__bf16)vn[jj][e];
    }
    __syncthreads();

    // ---- o = (rk*dexp)@S + intra@v_new (gated, to global); S' frags
    floatx8 sacc[2];
#pragma unroll
    for (int jj = 0; jj < 2; ++jj) {
      int ni = (nh * 2 + jj) * 16;
      floatx8 o_ = {};
      o_ = wmma_bf16(frag_ld(tl1, (mi >> 4) * 2 + 0, lane),
                     frag_ld(Sb, 0 + (ni >> 4), lane), o_);
      o_ = wmma_bf16(frag_ld(tl1, (mi >> 4) * 2 + 1, lane),
                     frag_ld(Sb, 4 + (ni >> 4), lane), o_);
      o_ = wmma_bf16(frag_ld(tl2, (mi >> 4) * 2 + 0, lane),
                     frag_ld(vnb, 0 + (ni >> 4), lane), o_);
      o_ = wmma_bf16(frag_ld(tl2, (mi >> 4) * 2 + 1, lane),
                     frag_ld(vnb, 4 + (ni >> 4), lane), o_);
      int ncol = ni + (lane & 15), mb = mi + ((lane >> 4) << 3);
#pragma unroll
      for (int e = 0; e < 8; ++e) {
        int m = mb + e, tt = n * Cc + m;
        float gv = gate[((size_t)(b * Tc + tt)) * Hc + h];
        og[((size_t)(b * Tc + tt)) * Dc + h * HDc + ncol] = o_[e] * gv;
      }
      floatx8 cfr = load_c_f(S, 64, mi, ni, lane);
#pragma unroll
      for (int e = 0; e < 8; ++e) cfr[e] *= elast;
      cfr = wmma_bf16(frag_ld(tl3, (mi >> 4) * 2 + 0, lane),
                      frag_ld(vnb, 0 + (ni >> 4), lane), cfr);
      cfr = wmma_bf16(frag_ld(tl3, (mi >> 4) * 2 + 1, lane),
                      frag_ld(vnb, 4 + (ni >> 4), lane), cfr);
      sacc[jj] = cfr;
    }
    __syncthreads();   // all reads of S / tiles done
#pragma unroll
    for (int jj = 0; jj < 2; ++jj) {
      int ni = (nh * 2 + jj) * 16;
      store_d_f(S, 64, mi, ni, lane, sacc[jj]);
    }
    __syncthreads();

    // ---- norm clip: scale = min(||S||,100)/max(||S||,1e-6)
    float ps = 0.f;
#pragma unroll
    for (int i = 0; i < 16; ++i) { float v = S[tid + i * 256]; ps += v * v; }
    part[tid] = ps;
    __syncthreads();
    for (int st = 128; st > 0; st >>= 1) {
      if (tid < st) part[tid] += part[tid + st];
      __syncthreads();
    }
    float nrm = sqrtf(part[0]);
    float scale = fminf(nrm, 100.f) / fmaxf(nrm, 1e-6f);
#pragma unroll
    for (int i = 0; i < 16; ++i) {
      int lin = tid + i * 256;
      float v = S[lin] * scale;
      S[lin] = v;
      Sb[bord64(lin >> 6, lin & 63)] = (__bf16)v;
    }
    __syncthreads();
  }
}

// ---------------------------------------------------------------------------
// Host-side launch
// ---------------------------------------------------------------------------
extern "C" void kernel_launch(void* const* d_in, const int* in_sizes, int n_in,
                              void* d_out, int out_size, void* d_ws, size_t ws_size,
                              hipStream_t stream) {
  const float* x       = (const float*)d_in[0];
  const float* w_write = (const float*)d_in[1];
  const float* w_gate  = (const float*)d_in[2];
  const float* w_out   = (const float*)d_in[3];
  const float* w_beta  = (const float*)d_in[4];
  const float* w_alpha = (const float*)d_in[5];
  const float* dt_bias = (const float*)d_in[6];
  const float* A_log   = (const float*)d_in[7];
  float* out = (float*)d_out;

  const size_t MD  = (size_t)Bc * Tc * Dc;
  const size_t BTH = (size_t)Bc * Tc * Hc;

  char* w = (char*)d_ws;
  auto alloc = [&](size_t bytes) -> void* {
    void* r = (void*)w;
    w += (bytes + 255) & ~(size_t)255;
    return r;
  };
  float*  xv    = (float*)alloc(MD * 4);
  float*  gateb = (float*)alloc(BTH * 4);
  float*  betab = (float*)alloc(BTH * 4);
  float*  decay = (float*)alloc(BTH * 4);
  float*  rk    = (float*)alloc(MD * 4);
  float*  wkA   = (float*)alloc(MD * 4);
  float*  vb    = (float*)alloc(MD * 4);
  float*  wkb   = (float*)alloc(MD * 4);
  float*  gcum  = (float*)alloc(BTH * 4);
  float*  vcorr = (float*)alloc(MD * 4);
  __bf16* t0g   = (__bf16*)alloc(MD * 2);
  __bf16* t1g   = (__bf16*)alloc(MD * 2);
  __bf16* t2g   = (__bf16*)alloc(MD * 2);
  __bf16* t3g   = (__bf16*)alloc(MD * 2);
  float*  og    = (float*)alloc(MD * 4);

  dim3 gemmGrid(Dc / 64, (Bc * Tc) / 64);
  gemm_bf16_kernel<<<gemmGrid, 256, 0, stream>>>(x, w_write, nullptr, xv,
                                                 Bc * Tc, Dc, Dc);
  smallproj_kernel<<<(unsigned)((BTH + 255) / 256), 256, 0, stream>>>(
      x, w_gate, w_beta, w_alpha, dt_bias, A_log, gateb, betab, decay);
  prep_kernel<<<Bc * Hc * Nc, 64, 0, stream>>>(x, xv, betab, decay,
                                               rk, wkA, vb, wkb, gcum);
  chunk_kernel<<<Bc * Hc * Nc, 256, 0, stream>>>(rk, wkA, vb, wkb, gcum,
                                                 vcorr, t0g, t1g, t2g, t3g);
  scan_kernel<<<Bc * Hc, 256, 0, stream>>>(vcorr, t0g, t1g, t2g, t3g, gcum,
                                           gateb, og);
  gemm_bf16_kernel<<<gemmGrid, 256, 0, stream>>>(og, w_out, x, out,
                                                 Bc * Tc, Dc, Dc);
}
